// DAS_SS2D_17849884082719
// MI455X (gfx1250) — matrix operand order repaired and checked by reading, not compiled, not run on hardware
//
#include <hip/hip_runtime.h>
#include <math.h>

#define DM   96            // D_MODEL
#define DI   192           // D_INNER
#define NST  16            // D_STATE
#define DTR  6             // DT_RANK
#define KD   4             // K_DIRS
#define GRP  8             // GROUP
#define BB   4
#define HH   96
#define WW   96
#define LL   (HH*WW)       // 9216
#define CDBL (DTR + 2*NST) // 38

typedef __attribute__((ext_vector_type(16))) _Float16 v16h;
typedef __attribute__((ext_vector_type(8)))  float    v8f;

// scan-order t -> physical flat index into (H*W) for direction k
__device__ __forceinline__ int map_phys(int k, int t) {
  switch (k & 3) {
    case 0:  return t;
    case 1:  return (t % WW) * WW + (t / WW);          // wh order
    case 2:  return (LL - 1) - t;                      // flipped
    default: { int u = (LL - 1) - t; return (u % WW) * WW + (u / WW); }
  }
}

// ---------------- K1: in_proj GEMM  xz = xh @ W^T  -> x1 (B,L,192), z (B,L,192)
__global__ void k_inproj(const float* __restrict__ x, const float* __restrict__ w,
                         float* __restrict__ x1, float* __restrict__ z) {
  const int lane = threadIdx.x & 31;
  const int wave = blockIdx.x * (blockDim.x >> 5) + (threadIdx.x >> 5);
  const int Nt = (2 * DI) / 16;           // 24 tiles along N=384
  const int tn = wave % Nt, tm = wave / Nt;
  const int half = lane >> 4;
  const int mrow = lane & 15, ncol = lane & 15;
  v8f acc = {};
  for (int k0 = 0; k0 < DM; k0 += 32) {
    v16h a, bf;
    const int mg = tm * 16 + mrow;
    const int b = mg / LL, hw = mg % LL;
#pragma unroll
    for (int e = 0; e < 16; ++e) {
      int kk = (e < 8) ? (half * 8 + e) : (8 + half * 8 + e);
      a[e] = (_Float16)x[(b * DM + (k0 + kk)) * LL + hw];
    }
#pragma unroll
    for (int e = 0; e < 16; ++e) {
      int kk = (e < 8) ? (half * 8 + e) : (8 + half * 8 + e);
      bf[e] = (_Float16)w[(tn * 16 + ncol) * DM + (k0 + kk)];
    }
    acc = __builtin_amdgcn_wmma_f32_16x16x32_f16(false, a, false, bf, (short)0, acc, false, false);
  }
#pragma unroll
  for (int r = 0; r < 8; ++r) {
    int m = tm * 16 + r + half * 8;
    int n = tn * 16 + ncol;
    float v = acc[r];
    if (n < DI) x1[m * DI + n]        = v;
    else        z [m * DI + (n - DI)] = v;
  }
}

// ---------------- K2: depthwise 3x3 conv + bias + SiLU, channel-last -> xc (B,L,192)
__global__ void k_dwconv(const float* __restrict__ x1, const float* __restrict__ cw,
                         const float* __restrict__ cb, float* __restrict__ xc) {
  const int c = threadIdx.x;                 // 0..191
  const int pix = blockIdx.x;                // b*L + hw
  const int b = pix / LL, hw = pix % LL;
  const int hy = hw / WW, wx = hw % WW;
  float s = cb[c];
#pragma unroll
  for (int i = 0; i < 3; ++i) {
    int yy = hy - 1 + i;
    if (yy < 0 || yy >= HH) continue;
#pragma unroll
    for (int j = 0; j < 3; ++j) {
      int xx = wx - 1 + j;
      if (xx < 0 || xx >= WW) continue;
      s += cw[c * 9 + i * 3 + j] * x1[(b * LL + yy * WW + xx) * DI + c];
    }
  }
  xc[pix * DI + c] = s / (1.f + __expf(-s));
}

// ---------------- K3: global avg + max pool over L per (b,c) -> pool (B,384)
__global__ void k_pool(const float* __restrict__ xc, float* __restrict__ pool) {
  const int d = threadIdx.x, b = blockIdx.x;
  float s = 0.f, mx = -3.4e38f;
  for (int hw = 0; hw < LL; ++hw) {
    float v = xc[(b * LL + hw) * DI + d];
    s += v; mx = fmaxf(mx, v);
  }
  pool[b * (2 * DI) + d]      = s / (float)LL;
  pool[b * (2 * DI) + DI + d] = mx;
}

// ---------------- K4: filt = pool @ conv_w^T ; filt_w = tanh ; prompt = filt(b,4,8)@emb
__global__ void k_filt(const float* __restrict__ pool, const float* __restrict__ cw,
                       const float* __restrict__ emb, float* __restrict__ filtw,
                       float* __restrict__ prompt) {
  __shared__ float sf[BB * 32];
  const int tid = threadIdx.x;
  if (tid < BB * 32) {
    int b = tid >> 5, j = tid & 31;
    float s = 0.f;
    for (int c = 0; c < 2 * DI; ++c) s += pool[b * (2 * DI) + c] * cw[j * (2 * DI) + c];
    sf[tid] = s;
    filtw[tid] = tanhf(s);                   // raw layout; consumer reads [b,g*4+k]
  }
  __syncthreads();
  if (tid < BB * KD * NST) {                 // 256
    int b = tid / (KD * NST), r = tid % (KD * NST);
    int k = r / NST, n = r % NST;
    float s = 0.f;
#pragma unroll
    for (int g = 0; g < GRP; ++g) s += sf[b * 32 + k * GRP + g] * emb[g * NST + n];
    prompt[tid] = s;
  }
}

// ---------------- K5: x_dbl GEMM per (b,k): [38x192] @ xs[192xL] via direction maps
__global__ void k_xdbl(const float* __restrict__ xc, const float* __restrict__ xpw,
                       float* __restrict__ xdbl) {
  const int lane = threadIdx.x & 31;
  const int wave = blockIdx.x * (blockDim.x >> 5) + (threadIdx.x >> 5);
  const int Nt = 3;                          // 38 -> 3 tiles of 16
  const int MtBK = LL / 16;                  // 576
  const int tn = wave % Nt;
  const int tmg = wave / Nt;
  const int bk = tmg / MtBK, tm = tmg % MtBK;
  const int b = bk >> 2, k = bk & 3;
  const int half = lane >> 4;
  const int mrow = lane & 15, ncol = lane & 15;
  const int t = tm * 16 + mrow;
  const int phys = map_phys(k, t);
  v8f acc = {};
  for (int k0 = 0; k0 < DI; k0 += 32) {
    v16h a, bf;
#pragma unroll
    for (int e = 0; e < 16; ++e) {
      int kk = (e < 8) ? (half * 8 + e) : (8 + half * 8 + e);
      a[e] = (_Float16)xc[(b * LL + phys) * DI + (k0 + kk)];
    }
#pragma unroll
    for (int e = 0; e < 16; ++e) {
      int kk = (e < 8) ? (half * 8 + e) : (8 + half * 8 + e);
      int c = tn * 16 + ncol;
      bf[e] = (_Float16)((c < CDBL) ? xpw[(k * CDBL + c) * DI + (k0 + kk)] : 0.f);
    }
    acc = __builtin_amdgcn_wmma_f32_16x16x32_f16(false, a, false, bf, (short)0, acc, false, false);
  }
  const int c = tn * 16 + ncol;
  if (c < CDBL) {
#pragma unroll
    for (int r = 0; r < 8; ++r) {
      int tt = tm * 16 + r + half * 8;
      xdbl[(bk * CDBL + c) * LL + tt] = acc[r];
    }
  }
}

// ---------------- K6: selective scan: 16 lanes (states) per channel, ds_swizzle reduce
__global__ void k_scan(const float* __restrict__ xc, const float* __restrict__ xdbl,
                       const float* __restrict__ alog, const float* __restrict__ dsv,
                       const float* __restrict__ dtw, const float* __restrict__ dtb,
                       const float* __restrict__ prompt, float* __restrict__ ys) {
  const int tid = blockIdx.x * blockDim.x + threadIdx.x;
  const int ch = tid >> 4;                   // (b,k,d) channel 0..3071
  const int n  = tid & 15;                   // state index
  const int b = ch / (KD * DI);
  const int rem = ch % (KD * DI);
  const int k = rem / DI, d = rem % DI;
  const float A  = -__expf(alog[(k * DI + d) * NST + n]);
  const float Dc = dsv[k * DI + d];
  float w[DTR];
#pragma unroll
  for (int r = 0; r < DTR; ++r) w[r] = dtw[(k * DI + d) * DTR + r];
  const float bias = dtb[k * DI + d];
  const float pr = prompt[(b * KD + k) * NST + n];
  const float* xd = xdbl + (b * KD + k) * CDBL * LL;
  float* yo = ys + ((b * KD + k) * DI + d) * LL;
  float h = 0.f;
  for (int t = 0; t < LL; ++t) {
    const int phys = map_phys(k, t);
    const float xt = xc[(b * LL + phys) * DI + d];
    float dts = bias;
#pragma unroll
    for (int r = 0; r < DTR; ++r) dts += w[r] * xd[r * LL + t];
    const float delta = (dts > 20.f) ? dts : __logf(1.f + __expf(dts));   // softplus
    const float Bt = xd[(DTR + n) * LL + t];
    const float Ct = xd[(DTR + NST + n) * LL + t] + pr;
    h = __expf(delta * A) * h + (delta * Bt) * xt;
    float yv = h * Ct;
    // reduce over the 16-lane state group (group-of-32 xor swizzles, masks < 16)
    yv += __uint_as_float(__builtin_amdgcn_ds_swizzle(__float_as_uint(yv), 0x041F));
    yv += __uint_as_float(__builtin_amdgcn_ds_swizzle(__float_as_uint(yv), 0x081F));
    yv += __uint_as_float(__builtin_amdgcn_ds_swizzle(__float_as_uint(yv), 0x101F));
    yv += __uint_as_float(__builtin_amdgcn_ds_swizzle(__float_as_uint(yv), 0x201F));
    if (n == 0) yo[t] = yv + Dc * xt;
  }
}

// ---------------- K7: directional recombine + filt_w mix + LN + SiLU gate
__global__ void k_combine(const float* __restrict__ ys, const float* __restrict__ filtw,
                          const float* __restrict__ alpha, const float* __restrict__ beta,
                          const float* __restrict__ lnw, const float* __restrict__ lnb,
                          const float* __restrict__ z, float* __restrict__ yout) {
  __shared__ float s1[DI], s2[DI], smv[2];
  const int d = threadIdx.x;
  const int pix = blockIdx.x;
  const int b = pix / LL, lo = pix % LL;
  const float* Y = ys + (size_t)b * KD * DI * LL;
  const int m1 = (lo % WW) * WW + lo / WW;
  // y_sum = y1+y2+y3+y4 at (d, lo)
  float ysum = Y[(0 * DI + d) * LL + lo]
             + Y[(2 * DI + d) * LL + (LL - 1 - lo)]
             + Y[(1 * DI + d) * LL + m1]
             + Y[(3 * DI + d) * LL + (LL - 1 - m1)];
  // y_main via the (b,8,24,L)->(b,h,w,192) reshape permutation
  const int f = lo * DI + d;
  const int g = f / (24 * LL);
  const int r2 = f % (24 * LL);
  const int dd = r2 / LL;
  const int l2 = r2 % LL;
  const int dp = g * 24 + dd;
  const int m1b = (l2 % WW) * WW + l2 / WW;
  const float* fw = filtw + b * 32 + g * 4;         // tanh(filt[b, g*4+k])
  float ymain = Y[(0 * DI + dp) * LL + l2]              * fw[0]
              + Y[(2 * DI + dp) * LL + (LL - 1 - l2)]   * fw[1]
              + Y[(1 * DI + dp) * LL + m1b]             * fw[2]
              + Y[(3 * DI + dp) * LL + (LL - 1 - m1b)]  * fw[3];
  const float yv = alpha[d] * ymain + beta[d] * ysum;
  s1[d] = yv; s2[d] = yv * yv;
  __syncthreads();
  if (d == 0) {
    float sa = 0.f, sb = 0.f;
    for (int i = 0; i < DI; ++i) { sa += s1[i]; sb += s2[i]; }
    float mu = sa / DI;
    smv[0] = mu;
    smv[1] = rsqrtf(sb / DI - mu * mu + 1e-5f);
  }
  __syncthreads();
  const float nv = (yv - smv[0]) * smv[1] * lnw[d] + lnb[d];
  const float zv = z[pix * DI + d];
  yout[pix * DI + d] = nv * (zv / (1.f + __expf(-zv)));
}

// ---------------- K8: out_proj GEMM -> (B,96,H,W)
__global__ void k_outproj(const float* __restrict__ y, const float* __restrict__ w,
                          float* __restrict__ out) {
  const int lane = threadIdx.x & 31;
  const int wave = blockIdx.x * (blockDim.x >> 5) + (threadIdx.x >> 5);
  const int Nt = DM / 16;                    // 6
  const int tn = wave % Nt, tm = wave / Nt;
  const int half = lane >> 4;
  const int mrow = lane & 15, ncol = lane & 15;
  v8f acc = {};
  for (int k0 = 0; k0 < DI; k0 += 32) {
    v16h a, bf;
#pragma unroll
    for (int e = 0; e < 16; ++e) {
      int kk = (e < 8) ? (half * 8 + e) : (8 + half * 8 + e);
      a[e] = (_Float16)y[(tm * 16 + mrow) * DI + (k0 + kk)];
    }
#pragma unroll
    for (int e = 0; e < 16; ++e) {
      int kk = (e < 8) ? (half * 8 + e) : (8 + half * 8 + e);
      bf[e] = (_Float16)w[(tn * 16 + ncol) * DI + (k0 + kk)];
    }
    acc = __builtin_amdgcn_wmma_f32_16x16x32_f16(false, a, false, bf, (short)0, acc, false, false);
  }
#pragma unroll
  for (int r = 0; r < 8; ++r) {
    int m = tm * 16 + r + half * 8;
    int b = m / LL, l = m % LL;
    out[(b * DM + tn * 16 + ncol) * LL + l] = acc[r];
  }
}

extern "C" void kernel_launch(void* const* d_in, const int* in_sizes, int n_in,
                              void* d_out, int out_size, void* d_ws, size_t ws_size,
                              hipStream_t stream) {
  const float* x    = (const float*)d_in[0];
  const float* ipw  = (const float*)d_in[1];
  const float* c2w  = (const float*)d_in[2];
  const float* c2b  = (const float*)d_in[3];
  const float* xpw  = (const float*)d_in[4];
  const float* dtw  = (const float*)d_in[5];
  const float* dtb  = (const float*)d_in[6];
  const float* alog = (const float*)d_in[7];
  const float* Dsp  = (const float*)d_in[8];
  const float* cw   = (const float*)d_in[9];
  const float* alp  = (const float*)d_in[10];
  const float* bet  = (const float*)d_in[11];
  const float* emb  = (const float*)d_in[12];
  const float* lnw  = (const float*)d_in[13];
  const float* lnb  = (const float*)d_in[14];
  const float* opw  = (const float*)d_in[15];

  float* ws = (float*)d_ws;
  const size_t NX  = (size_t)BB * LL * DI;          // 7,077,888
  const size_t NXD = (size_t)BB * KD * CDBL * LL;   // 5,603,328
  const size_t NYS = (size_t)BB * KD * DI * LL;     // 28,311,552
  float* x1     = ws;                // reused as yout after conv is consumed
  float* z      = ws + NX;
  float* xc     = ws + 2 * NX;
  float* xdbl   = ws + 3 * NX;
  float* ys     = xdbl + NXD;
  float* pool   = ys + NYS;
  float* filtw  = pool + BB * 2 * DI;
  float* prompt = filtw + BB * 32;
  float* yout   = x1;

  k_inproj <<<6912,    256, 0, stream>>>(x, ipw, x1, z);          // 55296 tiles / 8 waves
  k_dwconv <<<BB * LL, DI,  0, stream>>>(x1, c2w, c2b, xc);
  k_pool   <<<BB,      DI,  0, stream>>>(xc, pool);
  k_filt   <<<1,       256, 0, stream>>>(pool, cw, emb, filtw, prompt);
  k_xdbl   <<<3456,    256, 0, stream>>>(xc, xpw, xdbl);          // 27648 tiles
  k_scan   <<<192,     256, 0, stream>>>(xc, xdbl, alog, Dsp, dtw, dtb, prompt, ys);
  k_combine<<<BB * LL, DI,  0, stream>>>(ys, filtw, alp, bet, lnw, lnb, z, yout);
  k_outproj<<<1728,    256, 0, stream>>>(yout, opw, (float*)d_out); // 13824 tiles
}
